// SparseMeshPool_3719441678807
// MI455X (gfx1250) — matrix operand order repaired and checked by reading, not compiled, not run on hardware
//
#include <hip/hip_runtime.h>
#include <math.h>

#define C_DIM   256
#define HID     128
#define NTILES  8     // HID / 16
#define KCHUNKS 8     // C_DIM / 32
#define EPSV    1e-8f

typedef __attribute__((ext_vector_type(16))) __bf16 v16bf;
typedef __attribute__((ext_vector_type(16))) float  v16f;
typedef __attribute__((ext_vector_type(8)))  float  v8f;

// float -> bf16 using the compiler's native conversion (RNE; packed cvt where
// the target has it).
static __device__ __forceinline__ unsigned short f2bf(float f) {
    __bf16 b = (__bf16)f;
    return __builtin_bit_cast(unsigned short, b);
}

static __device__ __forceinline__ float gelu_exact(float x) {
    return 0.5f * x * (1.0f + erff(x * 0.70710678118654752f));
}

// ---------------------------------------------------------------------------
// Kernel 1: gate MLP.  gates[m] = sigmoid(gelu(x[m,:]@W1 + b1)@W2 + b2)
// 256 threads = 8 waves; wave w owns M-tile rows [blk*128 + w*16, +16).
// W1 staged in LDS as bf16, pre-swizzled into WMMA-B fragment order so each
// B fragment is one aligned 32B LDS read per lane.
// ---------------------------------------------------------------------------
__global__ __launch_bounds__(256) void gates_kernel(
    const float* __restrict__ x,  const float* __restrict__ W1,
    const float* __restrict__ b1, const float* __restrict__ W2,
    const float* __restrict__ b2, float* __restrict__ gates, int n_in)
{
    __shared__ alignas(32) unsigned short ldsW[KCHUNKS * NTILES * 32 * 16]; // 64 KB

    const int tid = threadIdx.x;

    // Coalesced fill: walk W1 linearly (float4 per thread per step; fully
    // coalesced global_load_b128), scatter bf16 into the swizzled B-fragment
    // layout on the LDS side.  This code's contiguous k order (A matches):
    //   lane = N%16 + 16*(K%32 / 16), element e = K%16.
    for (int base = tid * 4; base < C_DIM * HID; base += 256 * 4) {
        float4 w = *(const float4*)(W1 + base);
        int k  = base >> 7;          // W1 row   (K index, 0..255)
        int j  = base & 127;         // W1 col   (N index, 4-aligned)
        int kc = k >> 5;
        int kr = k & 31;
        int hf = kr >> 4;
        int e  = kr & 15;
        int nt = j >> 4;             // j..j+3 stay inside one 16-col block
        int n0 = j & 15;
        unsigned fragbase = (unsigned)(((kc * NTILES + nt) * 32 + 16 * hf + n0) * 16 + e);
        ldsW[fragbase]      = f2bf(w.x);
        ldsW[fragbase + 16] = f2bf(w.y);   // next lane -> +16 ushorts (32B)
        ldsW[fragbase + 32] = f2bf(w.z);
        ldsW[fragbase + 48] = f2bf(w.w);
    }
    __syncthreads();

    const int wave = tid >> 5;
    const int lane = tid & 31;
    const int hf   = lane >> 4;
    const int m0   = blockIdx.x * 128 + wave * 16;
    if (m0 >= n_in) return;

    // Accumulators initialized with bias b1 (C layout: N = lane%16, same for
    // all 8 rows held in the fragment).
    v8f acc[NTILES];
    for (int nt = 0; nt < NTILES; ++nt) {
        float bv = b1[nt * 16 + (lane & 15)];
        for (int e = 0; e < 8; ++e) acc[nt][e] = bv;
    }

    const int    mrow   = m0 + (lane & 15);
    const int    mclamp = (mrow < n_in) ? mrow : (n_in - 1);
    const float* xrow   = x + (size_t)mclamp * C_DIM;

    for (int kc = 0; kc < KCHUNKS; ++kc) {
        // A fragment (contiguous k order, as B): lane holds row M=lane%16,
        // K = e + 16*half + kc*32 -> two contiguous 8-float runs.
        const int k0 = kc * 32 + 16 * hf;
        const float4* p0 = (const float4*)(xrow + k0);
        const float4* p1 = (const float4*)(xrow + k0 + 8);
        float4 f0 = p0[0], f1 = p0[1];
        float4 f2 = p1[0], f3 = p1[1];
        v16f af;
        af[0]  = f0.x; af[1]  = f0.y; af[2]  = f0.z; af[3]  = f0.w;
        af[4]  = f1.x; af[5]  = f1.y; af[6]  = f1.z; af[7]  = f1.w;
        af[8]  = f2.x; af[9]  = f2.y; af[10] = f2.z; af[11] = f2.w;
        af[12] = f3.x; af[13] = f3.y; af[14] = f3.z; af[15] = f3.w;
        v16bf A = __builtin_convertvector(af, v16bf);

        for (int nt = 0; nt < NTILES; ++nt) {
            const v16bf* bp =
                (const v16bf*)&ldsW[((kc * NTILES + nt) * 32 + lane) * 16];
            acc[nt] = __builtin_amdgcn_wmma_f32_16x16x32_bf16(
                false, A, false, *bp, (short)0, acc[nt], false, false);
        }
    }

    // Second linear: per-lane partial dot with W2, reduce across each 16-lane
    // half (C layout: VGPR v, lanes 0-15 -> M=v; lanes 16-31 -> M=v+8).
    float w2v[NTILES];
    for (int nt = 0; nt < NTILES; ++nt) w2v[nt] = W2[nt * 16 + (lane & 15)];

    float partial[8];
    for (int v = 0; v < 8; ++v) {
        float s = 0.0f;
        for (int nt = 0; nt < NTILES; ++nt)
            s += gelu_exact(acc[nt][v]) * w2v[nt];
        partial[v] = s;
    }
    for (int off = 1; off < 16; off <<= 1)
        for (int v = 0; v < 8; ++v)
            partial[v] += __shfl_xor(partial[v], off, 32);

    if ((lane & 15) == 0) {
        float bb = b2[0];
        for (int v = 0; v < 8; ++v) {
            int m = m0 + hf * 8 + v;
            if (m < n_in) {
                float z = partial[v] + bb;
                gates[m] = 1.0f / (1.0f + expf(-z));
            }
        }
    }
}

// ---------------------------------------------------------------------------
// Kernel 2: row_ptr[r] = lower_bound(rows, r) over the sorted rows array.
// ---------------------------------------------------------------------------
__global__ void rowptr_kernel(const int* __restrict__ rows, int nnz, int n_out,
                              int* __restrict__ rowptr)
{
    int r = blockIdx.x * blockDim.x + threadIdx.x;
    if (r > n_out) return;
    if (r == n_out) { rowptr[n_out] = nnz; return; }
    int lo = 0, hi = nnz;
    while (lo < hi) {
        int mid = (lo + hi) >> 1;
        if (rows[mid] < r) lo = mid + 1; else hi = mid;
    }
    rowptr[r] = lo;
}

// ---------------------------------------------------------------------------
// Kernel 3: segmented gated mean-pool.  One wave per output row, 8 channels
// per lane held in registers, no atomics (rows are sorted).
// ---------------------------------------------------------------------------
__global__ __launch_bounds__(256) void pool_kernel(
    const float* __restrict__ x,     const float* __restrict__ vals,
    const int*   __restrict__ cols,  const float* __restrict__ gates,
    const int*   __restrict__ rowptr, float* __restrict__ out, int n_out)
{
    const int wave = threadIdx.x >> 5;
    const int lane = threadIdx.x & 31;
    const int r    = blockIdx.x * 8 + wave;
    if (r >= n_out) return;

    const int beg = rowptr[r];
    const int end = rowptr[r + 1];
    const int jb  = lane * 8;

    float a0=0.f,a1=0.f,a2=0.f,a3=0.f,a4=0.f,a5=0.f,a6=0.f,a7=0.f;
    float mass = 0.f;

    for (int e = beg; e < end; ++e) {
        int   c = cols[e];
        float g = vals[e] * gates[c];
        mass += g;
        const float4* xp = (const float4*)(x + (size_t)c * C_DIM + jb);
        float4 u = xp[0], w = xp[1];
        a0 += g * u.x; a1 += g * u.y; a2 += g * u.z; a3 += g * u.w;
        a4 += g * w.x; a5 += g * w.y; a6 += g * w.z; a7 += g * w.w;
    }

    float inv = 1.0f / fmaxf(mass, EPSV);
    float4* op = (float4*)(out + (size_t)r * C_DIM + jb);
    op[0] = make_float4(a0 * inv, a1 * inv, a2 * inv, a3 * inv);
    op[1] = make_float4(a4 * inv, a5 * inv, a6 * inv, a7 * inv);
}

// ---------------------------------------------------------------------------
extern "C" void kernel_launch(void* const* d_in, const int* in_sizes, int n_in_cnt,
                              void* d_out, int out_size, void* d_ws, size_t ws_size,
                              hipStream_t stream)
{
    const float* x    = (const float*)d_in[0];
    const float* vals = (const float*)d_in[1];
    const int*   rows = (const int*)  d_in[2];
    const int*   cols = (const int*)  d_in[3];
    const float* W1   = (const float*)d_in[4];
    const float* b1   = (const float*)d_in[5];
    const float* W2   = (const float*)d_in[6];
    const float* b2   = (const float*)d_in[7];
    float* out = (float*)d_out;

    const int N_IN  = in_sizes[0] / C_DIM;
    const int NNZ   = in_sizes[1];
    const int N_OUT = out_size / C_DIM;

    float* gates  = (float*)d_ws;
    size_t goff   = (((size_t)N_IN * sizeof(float)) + 255u) & ~(size_t)255u;
    int*   rowptr = (int*)((char*)d_ws + goff);

    (void)n_in_cnt; (void)ws_size;

    gates_kernel<<<(N_IN + 127) / 128, 256, 0, stream>>>(x, W1, b1, W2, b2, gates, N_IN);
    rowptr_kernel<<<(N_OUT + 1 + 255) / 256, 256, 0, stream>>>(rows, NNZ, N_OUT, rowptr);
    pool_kernel<<<(N_OUT + 7) / 8, 256, 0, stream>>>(x, vals, cols, gates, rowptr, out, N_OUT);
}
